// BatchAwareHSauteUnit_57028575756933
// MI455X (gfx1250) — matrix operand order repaired and checked by reading, not compiled, not run on hardware
//
#include <hip/hip_runtime.h>
#include <hip/hip_bf16.h>

// ---------------------------------------------------------------------------
// Types for CDNA5 WMMA (wave32).  v16bf = 16 bf16 = 8 VGPRs, v8f = 8 VGPRs.
// ---------------------------------------------------------------------------
typedef unsigned short bf16_t;                                   // raw bf16 bits
typedef __bf16 v16bf __attribute__((ext_vector_type(16)));
typedef float  v8f   __attribute__((ext_vector_type(8)));

__device__ __forceinline__ bf16_t f2bf_bits(float f) {
  union { float f; unsigned u; } a; a.f = f;
  unsigned u = a.u;
  unsigned r = (u + 0x7FFFu + ((u >> 16) & 1u)) >> 16;           // round-nearest-even
  return (bf16_t)r;
}

__device__ __forceinline__ v8f wmma_bf16(v16bf a, v16bf b, v8f c) {
  // D(f32 16x16) = A(16x32 bf16) * B(32x16 bf16) + C
  return __builtin_amdgcn_wmma_f32_16x16x32_bf16(false, a, false, b,
                                                 (short)0, c, false, false);
}

union FragU { v16bf v; uint4 q[2]; };

#define ZV8 v8f{0.f,0.f,0.f,0.f,0.f,0.f,0.f,0.f}

// ---------------------------------------------------------------------------
// Generic WMMA GEMM:  C[M,N] = act(A[M,K] @ Bw[K,N] + bias[N])
//   A: fp32 or bf16 row-major.  Bw: bf16 row-major [K][N].
//   Block tile 128x256, 8 waves (2x4), each wave 64x64 (4x4 WMMA tiles).
//   Double-buffered LDS staging: global loads for tile k+1 overlap the
//   16-WMMA burst for tile k; one barrier per K-step.
// ---------------------------------------------------------------------------
template<bool A_BF16, bool C_BF16, bool DO_GELU>
__global__ __launch_bounds__(256) void gemm_wmma_kernel(
    const void* __restrict__ Av, const bf16_t* __restrict__ Bw,
    const float* __restrict__ bias, void* __restrict__ Cv,
    int M, int N, int K)
{
  __shared__ __align__(16) bf16_t As[2][128][40];   // [m][k], pad 8
  __shared__ __align__(16) bf16_t Bs[2][256][40];   // [n][k] (B transposed), pad 8

  const int tid  = threadIdx.x;
  const int lane = tid & 31;
  const int w    = tid >> 5;                     // wave 0..7
  const int wm   = w >> 2;                       // 0..1 -> 64-row slab
  const int wn   = w & 3;                        // 0..3 -> 64-col slab
  const int ml   = lane & 15;
  const int half = lane >> 4;                    // 0/1

  const int mBase = blockIdx.y * 128;
  const int nBase = blockIdx.x * 256;

  const float*  Af = (const float*)Av;
  const bf16_t* Ab = (const bf16_t*)Av;

  // staging thread roles
  const int ar = tid >> 3;                       // A row 0..31 (step 32)
  const int ac = (tid & 7) * 4;                  // A col group (4 elems)
  const int bk = tid >> 3;                       // B k-row 0..31
  const int bn = (tid & 7) * 32;                 // B col group (32 elems)

  v8f acc[4][4];
#pragma unroll
  for (int i = 0; i < 4; ++i)
#pragma unroll
    for (int j = 0; j < 4; ++j) acc[i][j] = ZV8;

  // register staging for double buffering
  float4 aF[4]; uint2 aB[4]; uint4 bD[4];

  auto gload = [&](int kt) {
    const int k0 = kt * 32;
#pragma unroll
    for (int i = 0; i < 4; ++i) {
      const long off = (long)(mBase + ar + 32 * i) * K + k0 + ac;
      if (A_BF16) aB[i] = *(const uint2*)(Ab + off);
      else        aF[i] = *(const float4*)(Af + off);
    }
    const long boff = (long)(k0 + bk) * N + nBase + bn;
#pragma unroll
    for (int i = 0; i < 4; ++i) bD[i] = *(const uint4*)(Bw + boff + 8 * i);
  };

  auto sstore = [&](int buf) {
#pragma unroll
    for (int i = 0; i < 4; ++i) {
      if (A_BF16) {
        *(uint2*)&As[buf][ar + 32 * i][ac] = aB[i];
      } else {
        As[buf][ar + 32 * i][ac + 0] = f2bf_bits(aF[i].x);
        As[buf][ar + 32 * i][ac + 1] = f2bf_bits(aF[i].y);
        As[buf][ar + 32 * i][ac + 2] = f2bf_bits(aF[i].z);
        As[buf][ar + 32 * i][ac + 3] = f2bf_bits(aF[i].w);
      }
    }
#pragma unroll
    for (int i = 0; i < 4; ++i) {
      const bf16_t* pp = (const bf16_t*)&bD[i];
#pragma unroll
      for (int j = 0; j < 8; ++j) Bs[buf][bn + 8 * i + j][bk] = pp[j];
    }
  };

  const int KT = K >> 5;                         // K / 32 (exact)
  gload(0);
  sstore(0);
  __syncthreads();

  for (int kt = 0; kt < KT; ++kt) {
    const int cur = kt & 1;
    if (kt + 1 < KT) gload(kt + 1);              // overlap with WMMA burst

    FragU afr[4], bfr[4];
#pragma unroll
    for (int mt = 0; mt < 4; ++mt) {
      const int row = wm * 64 + mt * 16 + ml;
      afr[mt].q[0] = *(const uint4*)&As[cur][row][half * 8];       // k = half*8 + j
      afr[mt].q[1] = *(const uint4*)&As[cur][row][16 + half * 8];  // k = 16+half*8+j
    }
#pragma unroll
    for (int nt = 0; nt < 4; ++nt) {
      const int col = wn * 64 + nt * 16 + ml;
      bfr[nt].q[0] = *(const uint4*)&Bs[cur][col][half * 8];       // k = half*8 + j
      bfr[nt].q[1] = *(const uint4*)&Bs[cur][col][16 + half * 8];
    }
#pragma unroll
    for (int mt = 0; mt < 4; ++mt)
#pragma unroll
      for (int nt = 0; nt < 4; ++nt)
        acc[mt][nt] = wmma_bf16(afr[mt].v, bfr[nt].v, acc[mt][nt]);

    if (kt + 1 < KT) sstore(cur ^ 1);            // fill the other buffer
    __syncthreads();
  }

  // ---- epilogue: bias (+GELU) and store ----
  float*  Cf = (float*)Cv;
  bf16_t* Cb = (bf16_t*)Cv;
#pragma unroll
  for (int mt = 0; mt < 4; ++mt)
#pragma unroll
    for (int nt = 0; nt < 4; ++nt) {
      const int col = nBase + wn * 64 + nt * 16 + ml;
      const float bv = bias[col];
#pragma unroll
      for (int r = 0; r < 8; ++r) {
        const int row = mBase + wm * 64 + mt * 16 + r + half * 8;
        float v = acc[mt][nt][r] + bv;
        if (DO_GELU) v = 0.5f * v * (1.0f + erff(v * 0.70710678118654752f));
        const long o = (long)row * N + col;
        if (C_BF16) Cb[o] = f2bf_bits(v); else Cf[o] = v;
      }
    }
}

// ---------------------------------------------------------------------------
// Fused self-attention for one (sequence, head): softmax(QK^T/8)V.
// grid = (12 heads, 256 sequences), block = 256 threads (8 waves).
// qkv: bf16 [32768, 2304] (Q | K | V each 768 cols, head dim 64).
// ---------------------------------------------------------------------------
__global__ __launch_bounds__(256) void attention_kernel(
    const bf16_t* __restrict__ qkv, bf16_t* __restrict__ attnO)
{
  __shared__ __align__(16) bf16_t Qs[128][72];
  __shared__ __align__(16) bf16_t Ks[128][72];    // Bs_t for QK^T: Ks[n=l][k=dim]
  __shared__ __align__(16) bf16_t Vt[64][136];    // Bs_t for PV:   Vt[n=dim][k=l]
  __shared__ __align__(16) bf16_t Ps[128][136];   // softmax probs (A for PV)
  __shared__ float Ss[128][129];                  // scores f32

  const int tid  = threadIdx.x;
  const int lane = tid & 31;
  const int w    = tid >> 5;
  const int ml   = lane & 15;
  const int half = lane >> 4;
  const int hd   = blockIdx.x;                    // head 0..11
  const int bt   = blockIdx.y;                    // sequence 0..255
  const long rowBase = (long)bt * 128;

  for (int i = tid; i < 128 * 64; i += 256) {
    const int l = i >> 6, dIdx = i & 63;
    const long base = (rowBase + l) * 2304 + hd * 64 + dIdx;
    Qs[l][dIdx]  = qkv[base];
    Ks[l][dIdx]  = qkv[base + 768];
    Vt[dIdx][l]  = qkv[base + 1536];
  }
  __syncthreads();

  // ---- S = Q @ K^T : wave w owns query rows [16w,16w+16), all 8 col tiles ----
  {
    v8f sc[8];
#pragma unroll
    for (int i = 0; i < 8; ++i) sc[i] = ZV8;
    const int row = w * 16 + ml;
#pragma unroll
    for (int k0 = 0; k0 < 64; k0 += 32) {
      FragU ua;
      ua.q[0] = *(const uint4*)&Qs[row][k0 + half * 8];
      ua.q[1] = *(const uint4*)&Qs[row][k0 + 16 + half * 8];
#pragma unroll
      for (int nt = 0; nt < 8; ++nt) {
        FragU ub;
        const int col = nt * 16 + ml;
        ub.q[0] = *(const uint4*)&Ks[col][k0 + half * 8];
        ub.q[1] = *(const uint4*)&Ks[col][k0 + 16 + half * 8];
        sc[nt] = wmma_bf16(ua.v, ub.v, sc[nt]);
      }
    }
#pragma unroll
    for (int nt = 0; nt < 8; ++nt)
#pragma unroll
      for (int r = 0; r < 8; ++r)
        Ss[w * 16 + r + half * 8][nt * 16 + ml] = sc[nt][r];
  }
  __syncthreads();

  // ---- row softmax with 1/sqrt(64) scale ----
  if (tid < 128) {
    float m = -1e30f;
    for (int c = 0; c < 128; ++c) m = fmaxf(m, Ss[tid][c] * 0.125f);
    float s = 0.f;
    for (int c = 0; c < 128; ++c) {
      const float e = __expf(Ss[tid][c] * 0.125f - m);
      Ss[tid][c] = e; s += e;
    }
    const float inv = 1.0f / s;
    for (int c = 0; c < 128; ++c) Ps[tid][c] = f2bf_bits(Ss[tid][c] * inv);
  }
  __syncthreads();

  // ---- O = P @ V : wave w owns rows [16w,16w+16), 4 col tiles (64 dims) ----
  {
    v8f oc[4];
#pragma unroll
    for (int i = 0; i < 4; ++i) oc[i] = ZV8;
    const int row = w * 16 + ml;
#pragma unroll
    for (int k0 = 0; k0 < 128; k0 += 32) {
      FragU ua;
      ua.q[0] = *(const uint4*)&Ps[row][k0 + half * 8];
      ua.q[1] = *(const uint4*)&Ps[row][k0 + 16 + half * 8];
#pragma unroll
      for (int nt = 0; nt < 4; ++nt) {
        FragU ub;
        const int col = nt * 16 + ml;
        ub.q[0] = *(const uint4*)&Vt[col][k0 + half * 8];
        ub.q[1] = *(const uint4*)&Vt[col][k0 + 16 + half * 8];
        oc[nt] = wmma_bf16(ua.v, ub.v, oc[nt]);
      }
    }
#pragma unroll
    for (int nt = 0; nt < 4; ++nt)
#pragma unroll
      for (int r = 0; r < 8; ++r) {
        const int l = w * 16 + r + half * 8;
        const int dIdx = nt * 16 + ml;
        attnO[(rowBase + l) * 768 + hd * 64 + dIdx] = f2bf_bits(oc[nt][r]);
      }
  }
}

// ---------------------------------------------------------------------------
// X[row] = LayerNorm(X[row] + R[row]) * g + b   (in place; one block per row)
// ---------------------------------------------------------------------------
__global__ __launch_bounds__(256) void residual_ln_kernel(
    float* __restrict__ X, const float* __restrict__ R,
    const float* __restrict__ g, const float* __restrict__ b)
{
  __shared__ float red[256];
  const long base = (long)blockIdx.x * 768;
  const int t = threadIdx.x;
  float y[3], s = 0.f;
#pragma unroll
  for (int i = 0; i < 3; ++i) {
    y[i] = X[base + t + 256 * i] + R[base + t + 256 * i];
    s += y[i];
  }
  red[t] = s; __syncthreads();
  for (int st = 128; st > 0; st >>= 1) { if (t < st) red[t] += red[t + st]; __syncthreads(); }
  const float mean = red[0] * (1.0f / 768.0f);
  __syncthreads();
  float vs = 0.f;
#pragma unroll
  for (int i = 0; i < 3; ++i) { const float d = y[i] - mean; vs += d * d; }
  red[t] = vs; __syncthreads();
  for (int st = 128; st > 0; st >>= 1) { if (t < st) red[t] += red[t + st]; __syncthreads(); }
  const float rstd = rsqrtf(red[0] * (1.0f / 768.0f) + 1e-5f);
#pragma unroll
  for (int i = 0; i < 3; ++i) {
    const int c = t + 256 * i;
    X[base + c] = (y[i] - mean) * rstd * g[c] + b[c];
  }
}

// ---------------------------------------------------------------------------
// Embedding gather: X = tok_emb[ids] + pos_emb[l]
// ---------------------------------------------------------------------------
__global__ __launch_bounds__(256) void embed_kernel(
    const int* __restrict__ ids, const float* __restrict__ tok,
    const float* __restrict__ pos, float* __restrict__ X)
{
  const long e = (long)blockIdx.x * 256 + threadIdx.x;   // < 25,165,824
  const int h = (int)(e % 768);
  const long r = e / 768;                                 // bt*128 + l
  const int l = (int)(r & 127);
  const int id = ids[r];
  X[e] = tok[(long)id * 768 + h] + pos[(long)l * 768 + h];
}

// fp32 -> bf16, straight copy
__global__ void conv_nt_kernel(const float* __restrict__ in,
                               bf16_t* __restrict__ out, long n)
{
  const long e = (long)blockIdx.x * 256 + threadIdx.x;
  if (e < n) out[e] = f2bf_bits(in[e]);
}

// fp32 [N][K] -> bf16 [K][N]  (for "@ W.T" GEMMs)
__global__ void conv_t_kernel(const float* __restrict__ in,
                              bf16_t* __restrict__ out, int N, int K)
{
  const long e = (long)blockIdx.x * 256 + threadIdx.x;
  if (e >= (long)N * K) return;
  const int  n = (int)(e % N);
  const long k = e / N;
  out[e] = f2bf_bits(in[(long)n * K + k]);
}

// ---------------------------------------------------------------------------
// Token pooling: tq = proj_b (s_embed == 0); softmax over L with mask; tc.
// One block per (b,t).
// ---------------------------------------------------------------------------
__global__ __launch_bounds__(256) void pool_kernel(
    const float* __restrict__ X, const float* __restrict__ proj_b,
    const int* __restrict__ mask, float* __restrict__ tc)
{
  __shared__ float sl[128];
  const int bt = blockIdx.x, t = threadIdx.x;
  const long base = (long)bt * 128 * 768;
  if (t < 128) {
    float acc = 0.f;
    const float* xr = X + base + (long)t * 768;
    for (int h = 0; h < 768; ++h) acc += proj_b[h] * xr[h];
    sl[t] = mask[bt * 128 + t] ? acc : -1e30f;
  }
  __syncthreads();
  if (t == 0) {
    float m = -1e30f;
    for (int i = 0; i < 128; ++i) m = fmaxf(m, sl[i]);
    float s = 0.f;
    for (int i = 0; i < 128; ++i) { const float e = __expf(sl[i] - m); sl[i] = e; s += e; }
    const float inv = 1.0f / s;
    for (int i = 0; i < 128; ++i) sl[i] *= inv;
  }
  __syncthreads();
  for (int h = t; h < 768; h += 256) {
    float acc = 0.f;
    for (int l = 0; l < 128; ++l) acc += sl[l] * X[base + (long)l * 768 + h];
    tc[(long)bt * 768 + h] = acc;
  }
}

// speaker_context constant: sc[h] = sum_j cs_qkv_b[2H+j]*cs_out_w[h][j] + cs_out_b[h]
__global__ void scvec_kernel(const float* __restrict__ cs_qkv_b,
                             const float* __restrict__ cs_out_w,
                             const float* __restrict__ cs_out_b,
                             float* __restrict__ sc)
{
  for (int h = threadIdx.x; h < 768; h += 256) {
    float acc = cs_out_b[h];
    const float* wr = cs_out_w + (long)h * 768;
    const float* vb = cs_qkv_b + 1536;
    for (int j = 0; j < 768; ++j) acc += vb[j] * wr[j];
    sc[h] = acc;
  }
}

// gv[h] = sum_j sc[j]*gate_w[l][2H+j][h] + gate_b[l][h]
__global__ void gvvec_kernel(const float* __restrict__ sc,
                             const float* __restrict__ gate_w_l,
                             const float* __restrict__ gate_b_l,
                             float* __restrict__ gv)
{
  for (int h = threadIdx.x; h < 768; h += 256) {
    float acc = gate_b_l[h];
    for (int j = 0; j < 768; ++j) acc += sc[j] * gate_w_l[(long)(1536 + j) * 768 + h];
    gv[h] = acc;
  }
}

// updated[bt] = LN(tc[bt] @ gate_w[:768] + gv) with sln; one block per bt
__global__ __launch_bounds__(256) void gate_ln_kernel(
    const float* __restrict__ tc, const float* __restrict__ gate_w_l,
    const float* __restrict__ gv, const float* __restrict__ sg,
    const float* __restrict__ sb, float* __restrict__ upd)
{
  __shared__ float red[256];
  __shared__ float tcs[768];
  const int bt = blockIdx.x, t = threadIdx.x;
  for (int j = t; j < 768; j += 256) tcs[j] = tc[(long)bt * 768 + j];
  __syncthreads();
  float y[3], s = 0.f;
#pragma unroll
  for (int i = 0; i < 3; ++i) {
    const int h = t + 256 * i;
    float acc = gv[h];
    for (int j = 0; j < 768; ++j) acc += tcs[j] * gate_w_l[(long)j * 768 + h];
    y[i] = acc; s += acc;
  }
  red[t] = s; __syncthreads();
  for (int st = 128; st > 0; st >>= 1) { if (t < st) red[t] += red[t + st]; __syncthreads(); }
  const float mean = red[0] * (1.0f / 768.0f);
  __syncthreads();
  float vs = 0.f;
#pragma unroll
  for (int i = 0; i < 3; ++i) { const float d = y[i] - mean; vs += d * d; }
  red[t] = vs; __syncthreads();
  for (int st = 128; st > 0; st >>= 1) { if (t < st) red[t] += red[t + st]; __syncthreads(); }
  const float rstd = rsqrtf(red[0] * (1.0f / 768.0f) + 1e-5f);
#pragma unroll
  for (int i = 0; i < 3; ++i) {
    const int h = t + 256 * i;
    upd[(long)bt * 768 + h] = (y[i] - mean) * rstd * sg[h] + sb[h];
  }
}

// deterministic segment-sum: one block per speaker slot
__global__ void scatter_kernel(const float* __restrict__ upd,
                               const int* __restrict__ sid,
                               float* __restrict__ out_state)
{
  const int s = blockIdx.x;
  for (int h = threadIdx.x; h < 768; h += 256) {
    float acc = 0.f;
    for (int bt = 0; bt < 256; ++bt)
      if (sid[bt] == s) acc += upd[(long)bt * 768 + h];
    out_state[(long)s * 768 + h] = acc;
  }
}

// ---------------------------------------------------------------------------
// Orchestration
// ---------------------------------------------------------------------------
extern "C" void kernel_launch(void* const* d_in, const int* in_sizes, int n_in,
                              void* d_out, int out_size, void* d_ws, size_t ws_size,
                              hipStream_t stream) {
  (void)in_sizes; (void)n_in; (void)out_size; (void)ws_size;
  const int H = 768, H3 = 2304, I = 3072, NL = 2;
  const long M  = 8L * 32 * 128;           // 32768 rows
  const long XN = M * H;                   // 25,165,824 f32

  const int*   input_ids   = (const int*)d_in[0];
  const int*   attn_mask   = (const int*)d_in[1];
  const int*   speaker_ids = (const int*)d_in[2];
  const float* tok_emb     = (const float*)d_in[4];
  const float* pos_emb     = (const float*)d_in[5];
  const float* qkv_w       = (const float*)d_in[6];
  const float* qkv_b       = (const float*)d_in[7];
  const float* attn_out_w  = (const float*)d_in[8];
  const float* attn_out_b  = (const float*)d_in[9];
  const float* ffn_w1      = (const float*)d_in[10];
  const float* ffn_b1      = (const float*)d_in[11];
  const float* ffn_w2      = (const float*)d_in[12];
  const float* ffn_b2      = (const float*)d_in[13];
  const float* ln1_g       = (const float*)d_in[14];
  const float* ln1_b       = (const float*)d_in[15];
  const float* ln2_g       = (const float*)d_in[16];
  const float* ln2_b       = (const float*)d_in[17];
  const float* gate_w      = (const float*)d_in[18];
  const float* gate_b      = (const float*)d_in[19];
  const float* sln_g       = (const float*)d_in[20];
  const float* sln_b       = (const float*)d_in[21];
  const float* proj_b      = (const float*)d_in[23];
  const float* cs_qkv_b    = (const float*)d_in[25];
  const float* cs_out_w    = (const float*)d_in[26];
  const float* cs_out_b    = (const float*)d_in[27];

  float* X         = (float*)d_out;        // fp32 master activations
  float* out_state = (float*)d_out + XN;   // [S=8, 768]

  // workspace carve-up
  char* p = (char*)d_ws;
  auto carve = [&](size_t bytes) -> char* {
    char* r = p; p += (bytes + 255) & ~(size_t)255; return r;
  };
  bf16_t* Wqkv  = (bf16_t*)carve((size_t)NL * H3 * H * 2);
  bf16_t* Wout  = (bf16_t*)carve((size_t)NL * H  * H * 2);
  bf16_t* Wff1  = (bf16_t*)carve((size_t)NL * H  * I * 2);
  bf16_t* Wff2  = (bf16_t*)carve((size_t)NL * I  * H * 2);
  bf16_t* big   = (bf16_t*)carve((size_t)M * I * 2);   // qkv (M*2304) / ffn_mid (M*3072), disjoint lifetimes
  bf16_t* attnO = (bf16_t*)carve((size_t)M * H * 2);
  float*  proj  = (float*) carve((size_t)M * H * 4);   // f32 residual branch buffer
  float*  tcb   = (float*) carve((size_t)256 * H * 4);
  float*  updb  = (float*) carve((size_t)256 * H * 4);
  float*  scv   = (float*) carve(768 * 4);
  float*  gvv   = (float*) carve(768 * 4);

  // one-time (per call) bf16 weight conversion
  for (int l = 0; l < NL; ++l) {
    long nqkv = (long)H3 * H;
    conv_t_kernel <<<(nqkv + 255) / 256, 256, 0, stream>>>(qkv_w + l * nqkv, Wqkv + l * nqkv, H3, H);
    long nout = (long)H * H;
    conv_t_kernel <<<(nout + 255) / 256, 256, 0, stream>>>(attn_out_w + l * nout, Wout + l * nout, H, H);
    long nff = (long)H * I;
    conv_nt_kernel<<<(nff + 255) / 256, 256, 0, stream>>>(ffn_w1 + l * nff, Wff1 + l * nff, nff);
    conv_nt_kernel<<<(nff + 255) / 256, 256, 0, stream>>>(ffn_w2 + l * nff, Wff2 + l * nff, nff);
  }

  // embeddings
  embed_kernel<<<(unsigned)(XN / 256), 256, 0, stream>>>(input_ids, tok_emb, pos_emb, X);

  const unsigned MT = (unsigned)(M / 128);   // 256 row tiles
  for (int l = 0; l < NL; ++l) {
    // qkv = X @ qkv_w.T + qkv_b                     (fp32 in, bf16 out)
    gemm_wmma_kernel<false, true, false><<<dim3(H3 / 256, MT), 256, 0, stream>>>(
        X, Wqkv + (long)l * H3 * H, qkv_b + (long)l * H3, big, (int)M, H3, H);
    // per-head attention
    attention_kernel<<<dim3(12, 256), 256, 0, stream>>>(big, attnO);
    // proj = attnO @ out_w.T + out_b               (bf16 in, f32 out)
    gemm_wmma_kernel<true, false, false><<<dim3(H / 256, MT), 256, 0, stream>>>(
        attnO, Wout + (long)l * H * H, attn_out_b + (long)l * H, proj, (int)M, H, H);
    // X = LN(X + proj)
    residual_ln_kernel<<<(unsigned)M, 256, 0, stream>>>(X, proj, ln1_g + (long)l * H, ln1_b + (long)l * H);
    // mid = gelu(X @ W1 + b1)                      (fp32 in, bf16 out, gelu)
    gemm_wmma_kernel<false, true, true><<<dim3(I / 256, MT), 256, 0, stream>>>(
        X, Wff1 + (long)l * H * I, ffn_b1 + (long)l * I, big, (int)M, I, H);
    // proj = mid @ W2 + b2                         (bf16 in, f32 out)
    gemm_wmma_kernel<true, false, false><<<dim3(H / 256, MT), 256, 0, stream>>>(
        big, Wff2 + (long)l * I * H, ffn_b2 + (long)l * H, proj, (int)M, H, I);
    // X = LN(X + proj)
    residual_ln_kernel<<<(unsigned)M, 256, 0, stream>>>(X, proj, ln2_g + (long)l * H, ln2_b + (long)l * H);
  }

  // ---- tail (speaker-state path): only the last layer's result is returned ----
  const int lf = NL - 1;
  scvec_kernel  <<<1, 256, 0, stream>>>(cs_qkv_b, cs_out_w, cs_out_b, scv);
  gvvec_kernel  <<<1, 256, 0, stream>>>(scv, gate_w + (long)lf * H3 * H, gate_b + (long)lf * H, gvv);
  pool_kernel   <<<256, 256, 0, stream>>>(X, proj_b, attn_mask, tcb);
  gate_ln_kernel<<<256, 256, 0, stream>>>(tcb, gate_w + (long)lf * H3 * H, gvv,
                                          sln_g + (long)lf * H, sln_b + (long)lf * H, updb);
  scatter_kernel<<<8, 256, 0, stream>>>(updb, speaker_ids, out_state);
}
